// TaskClassifier_40621800686167
// MI455X (gfx1250) — matrix operand order repaired and checked by reading, not compiled, not run on hardware
//
#include <hip/hip_runtime.h>
#include <hip/hip_bf16.h>

// ---------------- problem constants ----------------
#define D_DIM    384
#define T_Y      11881
#define T_Y_PAD  11904          // 93 * 128
#define NP_SIDE  109
#define N_OBJ    4
#define T_OBJ    1024
#define M_TOTAL  (N_OBJ * T_OBJ)   // 4096
#define KSTEPS   (D_DIM / 32)      // 12

// LDS B-staging: 128 cols x 32 halfs, stride 40 halfs (80B: 64B data + 16B pad)
#define B_ROW_H  40
#define B_BUF_H  (128 * B_ROW_H)

typedef __attribute__((ext_vector_type(8)))  _Float16 v8h_t;
typedef __attribute__((ext_vector_type(16))) _Float16 v16h_t;
typedef __attribute__((ext_vector_type(8)))  float    v8f_t;

union HFrag { v16h_t v; v8h_t h[2]; };

// -------- async global->LDS copy (ASYNCcnt-tracked, CDNA5) ----------
__device__ __forceinline__ void async_cp_b128(const _Float16* g, _Float16* l) {
    unsigned loff = (unsigned)(unsigned long long)l;   // low 32 bits = LDS offset
    asm volatile("global_load_async_to_lds_b128 %0, %1, off"
                 :: "v"(loff), "v"(g) : "memory");
}

// ---------------- kernel 1: zero the per-object argmax keys ----------------
__global__ void init_keys_kernel(unsigned long long* keys) {
    if (threadIdx.x < N_OBJ) keys[threadIdx.x] = 0ULL;
}

// ---------------- kernel 2/3: row-normalize f32 -> f16 ----------------
__global__ __launch_bounds__(256) void normalize_kernel(
        const float* __restrict__ src, _Float16* __restrict__ dst,
        int rows_in, int rows_pad) {
    const int lane = threadIdx.x & 31;
    const int wid  = threadIdx.x >> 5;
    const int row  = blockIdx.x * 8 + wid;
    if (row >= rows_pad) return;
    _Float16* drow = dst + (size_t)row * D_DIM;
    if (row >= rows_in) {
        for (int i = lane; i < D_DIM; i += 32) drow[i] = (_Float16)0.0f;
        return;
    }
    const float* srow = src + (size_t)row * D_DIM;
    float v[D_DIM / 32];
    float ss = 0.0f;
#pragma unroll
    for (int i = 0; i < D_DIM / 32; ++i) {
        v[i] = srow[lane + 32 * i];
        ss += v[i] * v[i];
    }
#pragma unroll
    for (int off = 16; off > 0; off >>= 1) ss += __shfl_xor(ss, off, 32);
    const float scale = 1.0f / fmaxf(sqrtf(ss), 1e-8f);
#pragma unroll
    for (int i = 0; i < D_DIM / 32; ++i)
        drow[lane + 32 * i] = (_Float16)(v[i] * scale);
}

// ---------------- kernel 4: fused WMMA GEMM + per-object max/argmax --------
// grid = (32 Mtiles, 93 Ntiles); block = 256 (8 waves).
// B strip (32K x 128N) staged global->LDS via double-buffered async copies;
// each wave: 16 M-rows x 128 N-cols, 12 K-steps of v_wmma_f32_16x16x32_f16.
__global__ __launch_bounds__(256) void gemm_argmax_kernel(
        const _Float16* __restrict__ onrm,   // [4096][384] normalized obj rows
        const _Float16* __restrict__ xnrm,   // [11904][384] normalized query rows
        unsigned long long* __restrict__ keys) {
    __shared__ __align__(16) _Float16 sB[2][B_BUF_H];   // 2 x 10240 B
    __shared__ unsigned long long smax;

    const int tid    = threadIdx.x;
    const int lane   = tid & 31;
    const int wid    = tid >> 5;
    const int l15    = lane & 15;
    const int half16 = lane >> 4;                 // 0 for lanes 0-15, 1 for 16-31

    if (tid == 0) smax = 0ULL;

    const int baseM = blockIdx.x * 128 + wid * 16;   // object-row tile for this wave
    const int baseN = blockIdx.y * 128;              // query-col tile for this block

    // Per-thread staging assignment: 512 chunks of 16B per K-step, 2 per thread.
    const int srow  = tid >> 2;        // 0..63
    const int spart = tid & 3;         // 0..3  (16B chunk within 64B row)
    const _Float16* gB0 = xnrm + (size_t)(baseN + srow) * D_DIM + spart * 8;
    const _Float16* gB1 = gB0 + (size_t)64 * D_DIM;

    // prefetch K-step 0 into buffer 0
    async_cp_b128(gB0, &sB[0][srow * B_ROW_H + spart * 8]);
    async_cp_b128(gB1, &sB[0][(srow + 64) * B_ROW_H + spart * 8]);

    const _Float16* aRow = onrm + (size_t)(baseM + l15) * D_DIM;

    v8f_t acc[8];
    const v8f_t vzero = {0.f, 0.f, 0.f, 0.f, 0.f, 0.f, 0.f, 0.f};
#pragma unroll
    for (int t = 0; t < 8; ++t) acc[t] = vzero;

    for (int step = 0; step < KSTEPS; ++step) {
        const int kk = step * 32;
        if (step + 1 < KSTEPS) {
            // issue next K-step's strip into the other buffer, then wait for
            // the current strip (async loads complete in order; <=2 pending
            // means the previous 2 are done)
            _Float16* nb = &sB[(step + 1) & 1][0];
            async_cp_b128(gB0 + kk + 32, nb + srow * B_ROW_H + spart * 8);
            async_cp_b128(gB1 + kk + 32, nb + (srow + 64) * B_ROW_H + spart * 8);
            asm volatile("s_wait_asynccnt 0x2" ::: "memory");
        } else {
            asm volatile("s_wait_asynccnt 0x0" ::: "memory");
        }
        __syncthreads();   // current buffer visible to all waves

        // A fragment: 16x32 f16, contiguous K layout matching B:
        // lane<16 holds K {kk..kk+15}; lane>=16 holds {kk+16..kk+31}
        HFrag a;
        a.h[0] = *(const v8h_t*)(aRow + kk + half16 * 16);
        a.h[1] = *(const v8h_t*)(aRow + kk + half16 * 16 + 8);

        const _Float16* bBase = &sB[step & 1][0] + half16 * 16;
#pragma unroll
        for (int t = 0; t < 8; ++t) {
            // B fragment: column (= query row) per lane pair, K contiguous
            // with split at 16: lane<16 -> K kk..kk+15, lane>=16 -> +16
            const _Float16* bRow = bBase + (t * 16 + l15) * B_ROW_H;
            HFrag bf;
            bf.h[0] = *(const v8h_t*)(bRow);
            bf.h[1] = *(const v8h_t*)(bRow + 8);
            acc[t] = __builtin_amdgcn_wmma_f32_16x16x32_f16(
                false, a.v, false, bf.v, (short)0, acc[t], false, false);
        }
        __syncthreads();   // all waves done reading before buffer is rewritten
    }

    // ---- epilogue: pack (ordered-float << 32) | ~flat_idx, take max ----
    // flat_idx = (m % 1024) * T_Y + n  (matches jnp.argmax flatten order);
    // ~flat gives first-occurrence tie-break under u64 max.
    unsigned long long best = 0ULL;
#pragma unroll
    for (int t = 0; t < 8; ++t) {
        const int n = baseN + t * 16 + l15;
        if (n < T_Y) {
#pragma unroll
            for (int r = 0; r < 8; ++r) {
                const int m = baseM + r + half16 * 8;
                unsigned int fb  = __float_as_uint(acc[t][r]);
                unsigned int ord = (fb & 0x80000000u) ? ~fb : (fb | 0x80000000u);
                unsigned int flat = (unsigned int)((m & (T_OBJ - 1)) * T_Y + n);
                unsigned long long key =
                    ((unsigned long long)ord << 32) | (unsigned long long)(~flat);
                if (key > best) best = key;
            }
        }
    }
    atomicMax(&smax, best);
    __syncthreads();
    if (tid == 0) {
        const int obj = (int)(blockIdx.x >> 3);   // 128-row tiles, 1024 rows/object
        atomicMax(&keys[obj], smax);
    }
}

// ---------------- kernel 5: decode + classifier + softmax ----------------
__global__ void finalize_kernel(const unsigned long long* __restrict__ keys,
                                const float* __restrict__ W,
                                const float* __restrict__ b,
                                float* __restrict__ out) {
    if (threadIdx.x != 0 || blockIdx.x != 0) return;
    float feat[N_OBJ * 3];
#pragma unroll
    for (int o = 0; o < N_OBJ; ++o) {
        const unsigned long long k = keys[o];
        const unsigned int ord = (unsigned int)(k >> 32);
        const unsigned int fb =
            (ord & 0x80000000u) ? (ord ^ 0x80000000u) : ~ord;
        const float val = __uint_as_float(fb);
        const unsigned int flat = ~((unsigned int)(k & 0xFFFFFFFFu));
        const int patch = (int)(flat % (unsigned int)T_Y);
        const int yd = patch / NP_SIDE;
        const int xd = patch % NP_SIDE;
        // (x-1)*STRIDE + STRIDE + PATCH/2 - 0.5 = 2*x + 3.5
        const float cx = (float)xd * 2.0f + 3.5f;
        const float cy = (float)yd * 2.0f + 3.5f;
        const bool found = (val >= 0.56f);
        feat[o * 3 + 0] = found ? cx : -99.0f;
        feat[o * 3 + 1] = found ? cy : -99.0f;
        feat[o * 3 + 2] = found ? 0.0f : -99.0f;
    }
    float logits[N_OBJ];
    float mx = -1e30f;
#pragma unroll
    for (int i = 0; i < N_OBJ; ++i) {
        float s = b[i];
#pragma unroll
        for (int j = 0; j < N_OBJ * 3; ++j) s += W[i * (N_OBJ * 3) + j] * feat[j];
        s = fmaxf(s, 0.0f);
        logits[i] = s;
        mx = fmaxf(mx, s);
    }
    float den = 0.0f;
    float e[N_OBJ];
#pragma unroll
    for (int i = 0; i < N_OBJ; ++i) { e[i] = expf(logits[i] - mx); den += e[i]; }
#pragma unroll
    for (int i = 0; i < N_OBJ; ++i) out[i] = e[i] / den;
}

// ---------------- host launch ----------------
extern "C" void kernel_launch(void* const* d_in, const int* in_sizes, int n_in,
                              void* d_out, int out_size, void* d_ws, size_t ws_size,
                              hipStream_t stream) {
    const float* x   = (const float*)d_in[0];   // [1,1,11881,384]
    const float* obj = (const float*)d_in[1];   // [4,1024,384]
    const float* W   = (const float*)d_in[2];   // [4,12]
    const float* b   = (const float*)d_in[3];   // [4]
    float* out = (float*)d_out;

    // workspace layout: [0,32) keys; [256, +3.0MB) on_f16; then xn_f16 (9.1MB)
    unsigned long long* keys = (unsigned long long*)d_ws;
    _Float16* onrm = (_Float16*)((char*)d_ws + 256);
    _Float16* xnrm = (_Float16*)((char*)d_ws + 256 + (size_t)M_TOTAL * D_DIM * 2);

    init_keys_kernel<<<1, 32, 0, stream>>>(keys);
    normalize_kernel<<<M_TOTAL / 8, 256, 0, stream>>>(obj, onrm, M_TOTAL, M_TOTAL);
    normalize_kernel<<<T_Y_PAD / 8, 256, 0, stream>>>(x, xnrm, T_Y, T_Y_PAD);

    dim3 grid(M_TOTAL / 128, T_Y_PAD / 128);    // 32 x 93
    gemm_argmax_kernel<<<grid, 256, 0, stream>>>(onrm, xnrm, keys);

    finalize_kernel<<<1, 32, 0, stream>>>(keys, W, b, out);
}
